// GATConv_pytm_22565758173852
// MI455X (gfx1250) — matrix operand order repaired and checked by reading, not compiled, not run on hardware
//
#include <hip/hip_runtime.h>
#include <hip/hip_bf16.h>

#define N_NODES  50000
#define N_EDGES  1600000
#define IN_FEATS 256
#define NUM_HEADS 4
#define OUT_F    32
#define HF       128           // NUM_HEADS * OUT_F
#define NEG_SLOPE 0.2f

typedef __attribute__((ext_vector_type(16))) __bf16 v16bf;
typedef __attribute__((ext_vector_type(8)))  float  v8f;

// ---------------------------------------------------------------------------
// 1) Projection GEMM: h[N,128] = feat[N,256] @ W_fc^T   (W_fc is [128,256])
//    bf16 WMMA 16x16x32, f32 accumulate. Block = 16 rows x 128 cols (8 waves).
// ---------------------------------------------------------------------------
__global__ __launch_bounds__(256) void gat_proj_wmma(
    const float* __restrict__ feat, const float* __restrict__ W,
    float* __restrict__ h)
{
    const int lane = threadIdx.x & 31;
    const int wave = threadIdx.x >> 5;          // 0..7 -> 16-col tile
    const int r    = lane & 15;
    const int hi   = lane >> 4;                 // lane half (0/1)
    const size_t rowBase = (size_t)blockIdx.x * 16;
    const int    colBase = wave * 16;

    const float* arow = feat + (rowBase + r) * IN_FEATS;         // A row M=r
    const float* bcol = W + (size_t)(colBase + r) * IN_FEATS;    // B col N=r

    v8f acc = {};
#pragma unroll
    for (int kc = 0; kc < IN_FEATS / 32; ++kc) {
        const int kb = kc * 32;
        // A (16x32 bf16): elem i<8 -> K=kb+hi*8+i ; i>=8 -> K=kb+16+hi*8+(i-8)
        const float4* ap = (const float4*)(arow + kb + hi * 8);
        float4 a0 = ap[0], a1 = ap[1], a2 = ap[4], a3 = ap[5];
        // B (32x16 bf16): elem i<8 -> K=kb+hi*8+i ; i>=8 -> K=kb+16+hi*8+(i-8)
        const float4* bp = (const float4*)(bcol + kb + hi * 8);
        float4 b0 = bp[0], b1 = bp[1], b2 = bp[4], b3 = bp[5];

        float af[16] = {a0.x,a0.y,a0.z,a0.w, a1.x,a1.y,a1.z,a1.w,
                        a2.x,a2.y,a2.z,a2.w, a3.x,a3.y,a3.z,a3.w};
        float bfv[16] = {b0.x,b0.y,b0.z,b0.w, b1.x,b1.y,b1.z,b1.w,
                         b2.x,b2.y,b2.z,b2.w, b3.x,b3.y,b3.z,b3.w};
        v16bf a, b;
#pragma unroll
        for (int i = 0; i < 16; ++i) { a[i] = (__bf16)af[i]; b[i] = (__bf16)bfv[i]; }

        acc = __builtin_amdgcn_wmma_f32_16x16x32_bf16(
            /*neg_a=*/false, a, /*neg_b=*/false, b,
            /*c_mod=*/(short)0, acc, /*reuse_a=*/false, /*reuse_b=*/false);
    }
    // D layout: elem j -> row = rowBase + j + hi*8, col = colBase + r
#pragma unroll
    for (int j = 0; j < 8; ++j)
        h[(rowBase + j + hi * 8) * HF + colBase + r] = acc[j];
}

// ---------------------------------------------------------------------------
// 2) Per-node attention logits: el/er[N,4] = sum_F h*attn  (one wave = 1 head)
// ---------------------------------------------------------------------------
__global__ __launch_bounds__(128) void gat_logits(
    const float* __restrict__ h, const float* __restrict__ attn_l,
    const float* __restrict__ attn_r, float* __restrict__ el,
    float* __restrict__ er)
{
    const int node = blockIdx.x;
    const int t = threadIdx.x;              // head = t/32, f = t%32
    float hv = h[(size_t)node * HF + t];
    float vl = hv * attn_l[t];
    float vr = hv * attn_r[t];
#pragma unroll
    for (int off = 16; off > 0; off >>= 1) {
        vl += __shfl_xor(vl, off, 32);
        vr += __shfl_xor(vr, off, 32);
    }
    if ((t & 31) == 0) {
        el[node * NUM_HEADS + (t >> 5)] = vl;
        er[node * NUM_HEADS + (t >> 5)] = vr;
    }
}

// ---------------------------------------------------------------------------
// Monotonic uint encoding so u32 atomicMax == float max
// ---------------------------------------------------------------------------
__device__ __forceinline__ unsigned int fenc(float f) {
    unsigned int u = __float_as_uint(f);
    return (u & 0x80000000u) ? ~u : (u | 0x80000000u);
}
__device__ __forceinline__ float fdec(unsigned int u) {
    return (u & 0x80000000u) ? __uint_as_float(u & 0x7FFFFFFFu)
                             : __uint_as_float(~u);
}

__global__ void gat_init(unsigned int* __restrict__ menc,
                         float* __restrict__ denom, float* __restrict__ out,
                         int nm, int nout)
{
    int i = blockIdx.x * blockDim.x + threadIdx.x;
    if (i < nm) { menc[i] = 0x007FFFFFu /* enc(-inf) */; denom[i] = 0.f; }
    if (i < nout) out[i] = 0.f;
}

// ---------------------------------------------------------------------------
// 3) Segment max over dst (per edge-head)
// ---------------------------------------------------------------------------
__global__ __launch_bounds__(256) void gat_edge_max(
    const int* __restrict__ src, const int* __restrict__ dst,
    const float* __restrict__ el, const float* __restrict__ er,
    unsigned int* __restrict__ menc)
{
    int i = blockIdx.x * blockDim.x + threadIdx.x;
    if (i >= N_EDGES * NUM_HEADS) return;
    int e = i >> 2, head = i & 3;
    float v = el[src[e] * NUM_HEADS + head] + er[dst[e] * NUM_HEADS + head];
    v = (v >= 0.f) ? v : NEG_SLOPE * v;
    atomicMax(&menc[dst[e] * NUM_HEADS + head], fenc(v));
}

__global__ void gat_m_fix(unsigned int* __restrict__ menc,
                          float* __restrict__ m, int n)
{
    int i = blockIdx.x * blockDim.x + threadIdx.x;
    if (i >= n) return;
    float f = fdec(menc[i]);
    m[i] = (f < -3.0e38f) ? 0.f : f;       // guard 0-in-degree (-inf -> 0)
}

// ---------------------------------------------------------------------------
// 4) Segment sum of exp(e - m[dst])
// ---------------------------------------------------------------------------
__global__ __launch_bounds__(256) void gat_edge_expsum(
    const int* __restrict__ src, const int* __restrict__ dst,
    const float* __restrict__ el, const float* __restrict__ er,
    const float* __restrict__ m, float* __restrict__ denom)
{
    int i = blockIdx.x * blockDim.x + threadIdx.x;
    if (i >= N_EDGES * NUM_HEADS) return;
    int e = i >> 2, head = i & 3;
    int d4 = dst[e] * NUM_HEADS + head;
    float v = el[src[e] * NUM_HEADS + head] + er[d4];
    v = (v >= 0.f) ? v : NEG_SLOPE * v;
    atomicAdd(&denom[d4], __expf(v - m[d4]));
}

// ---------------------------------------------------------------------------
// 5) Weighted scatter aggregation: out[dst] += h[src] * softmax-weight
//    block = 256 threads = 2 edges x 128 features
// ---------------------------------------------------------------------------
__global__ __launch_bounds__(256) void gat_edge_aggregate(
    const int* __restrict__ src, const int* __restrict__ dst,
    const float* __restrict__ el, const float* __restrict__ er,
    const float* __restrict__ m, const float* __restrict__ denom,
    const float* __restrict__ h, float* __restrict__ out)
{
    int e = blockIdx.x * 2 + (threadIdx.x >> 7);
    if (e >= N_EDGES) return;
    int c = threadIdx.x & 127;              // head = c/32, feature = c%32
    int head = c >> 5;
    int s = src[e], d = dst[e];
    int d4 = d * NUM_HEADS + head;
    float v = el[s * NUM_HEADS + head] + er[d4];
    v = (v >= 0.f) ? v : NEG_SLOPE * v;
    float dn = denom[d4];
    dn = (dn > 0.f) ? dn : 1.f;
    float a = __expf(v - m[d4]) / dn;
    atomicAdd(&out[(size_t)d * HF + c], h[(size_t)s * HF + c] * a);
}

// ---------------------------------------------------------------------------
// 6) bias add
// ---------------------------------------------------------------------------
__global__ void gat_bias(float* __restrict__ out,
                         const float* __restrict__ bias, int n)
{
    int i = blockIdx.x * blockDim.x + threadIdx.x;
    if (i < n) out[i] += bias[i & (HF - 1)];
}

// ---------------------------------------------------------------------------
extern "C" void kernel_launch(void* const* d_in, const int* in_sizes, int n_in,
                              void* d_out, int out_size, void* d_ws, size_t ws_size,
                              hipStream_t stream)
{
    const float* feat   = (const float*)d_in[0];
    const float* W_fc   = (const float*)d_in[1];
    const float* attn_l = (const float*)d_in[2];
    const float* attn_r = (const float*)d_in[3];
    const float* bias   = (const float*)d_in[4];
    const int*   src    = (const int*)d_in[5];
    const int*   dst    = (const int*)d_in[6];
    float* out = (float*)d_out;

    // workspace layout (floats): h[N*128] | el[N*4] | er[N*4] | m[N*4] | denom[N*4]
    float* h     = (float*)d_ws;
    float* el    = h  + (size_t)N_NODES * HF;
    float* er    = el + (size_t)N_NODES * NUM_HEADS;
    float* m     = er + (size_t)N_NODES * NUM_HEADS;
    float* denom = m  + (size_t)N_NODES * NUM_HEADS;
    unsigned int* menc = (unsigned int*)m;   // encoded during max phase

    const int NM   = N_NODES * NUM_HEADS;    // 200000
    const int NOUT = N_NODES * HF;           // 6400000
    const int NEH  = N_EDGES * NUM_HEADS;    // 6400000

    gat_proj_wmma<<<N_NODES / 16, 256, 0, stream>>>(feat, W_fc, h);
    gat_logits<<<N_NODES, 128, 0, stream>>>(h, attn_l, attn_r, el, er);
    gat_init<<<(NOUT + 255) / 256, 256, 0, stream>>>(menc, denom, out, NM, NOUT);
    gat_edge_max<<<(NEH + 255) / 256, 256, 0, stream>>>(src, dst, el, er, menc);
    gat_m_fix<<<(NM + 255) / 256, 256, 0, stream>>>(menc, m, NM);
    gat_edge_expsum<<<(NEH + 255) / 256, 256, 0, stream>>>(src, dst, el, er, m, denom);
    gat_edge_aggregate<<<N_EDGES / 2, 256, 0, stream>>>(src, dst, el, er, m, denom, h, out);
    gat_bias<<<(NOUT + 255) / 256, 256, 0, stream>>>(out, bias, NOUT);
}
